// InformerLiteBlock_43885975830986
// MI455X (gfx1250) — matrix operand order repaired and checked by reading, not compiled, not run on hardware
//
#include <hip/hip_runtime.h>
#include <math.h>

// ---------------- CDNA5 WMMA types ----------------
typedef __attribute__((ext_vector_type(16))) __bf16 v16bf;
typedef __attribute__((ext_vector_type(8)))  float  v8f;

union AF { v16bf v; unsigned int u[8]; unsigned short s[16]; };

#define D_MODEL 512
#define SEQ_T   768
#define BATCH   16
#define NHEAD   8
#define DKH     64
#define T_OUT   384
#define QKV_N   1536
#define CONV_K  1536

__device__ __forceinline__ unsigned short f2bf(float f) {
  unsigned int u = __float_as_uint(f);
  u += 0x7FFFu + ((u >> 16) & 1u);   // round-to-nearest-even
  return (unsigned short)(u >> 16);
}

__device__ __forceinline__ v8f wmma_bf16(v16bf a, v16bf b, v8f c) {
  // D = A(16x32 bf16) x B(32x16 bf16) + C(16x16 f32)
  return __builtin_amdgcn_wmma_f32_16x16x32_bf16(false, a, false, b, (short)0, c, false, false);
}

// ---------------- conversion kernels ----------------
__global__ void k_cvt(const float* __restrict__ in, unsigned short* __restrict__ out, int n) {
  int i = blockIdx.x * blockDim.x + threadIdx.x;
  if (i < n) out[i] = f2bf(in[i]);
}

// Wconv[dout][din][kk] -> Wc2[dout][kc = kk*512 + din]  (im2col weight layout)
__global__ void k_cvt_wconv(const float* __restrict__ w, unsigned short* __restrict__ wc, int n) {
  int i = blockIdx.x * blockDim.x + threadIdx.x;
  if (i < n) {
    int dout = i / CONV_K;
    int kc   = i % CONV_K;
    int kk   = kc >> 9;        // kc / 512
    int din  = kc & 511;       // kc % 512
    wc[i] = f2bf(w[dout * CONV_K + din * 3 + kk]);
  }
}

// ---------------- QKV GEMM: [12288,512] x Wqkv^T -> scatter Q,K,V^T (bf16) ----------------
// one wave computes a 16(M) x 64(N) strip; 768 M-tiles x 24 N-strips = 18432 waves
__global__ void k_gemm_qkv(const unsigned short* __restrict__ xb,
                           const unsigned short* __restrict__ wq,   // Wqkv bf16 [1536][512] (row-major, orig layout)
                           const float* __restrict__ bqkv,
                           unsigned short* __restrict__ Qb,         // [b,h,t,dk]
                           unsigned short* __restrict__ Kb,         // [b,h,t,dk]
                           unsigned short* __restrict__ Vt) {       // [b,h,dk,t]
  const int lane = threadIdx.x & 31;
  const int wg   = blockIdx.x * 8 + (threadIdx.x >> 5);
  const int mt   = wg / 24;
  const int ns   = wg % 24;
  const int m    = lane & 15, half = lane >> 4;

  const unsigned short* arow = xb + (mt * 16 + m) * D_MODEL;
  v8f acc[4] = {};

  for (int ks = 0; ks < D_MODEL / 32; ++ks) {
    AF A;
#pragma unroll
    for (int j = 0; j < 8; ++j) {
      int k = ks * 32 + ((j >> 2) << 4) + half * 8 + ((j & 3) << 1);
      A.u[j] = *(const unsigned int*)(arow + k);
    }
#pragma unroll
    for (int s = 0; s < 4; ++s) {
      const unsigned short* bcol = wq + (ns * 64 + s * 16 + m) * D_MODEL + ks * 32;
      AF Bf;
#pragma unroll
      for (int j = 0; j < 8; ++j)
        Bf.u[j] = *(const unsigned int*)(bcol + ((j >> 2) << 4) + half * 8 + ((j & 3) << 1));
      acc[s] = wmma_bf16(A.v, Bf.v, acc[s]);
    }
  }

#pragma unroll
  for (int s = 0; s < 4; ++s) {
#pragma unroll
    for (int r = 0; r < 8; ++r) {
      int mg = mt * 16 + half * 8 + r;
      int ng = ns * 64 + s * 16 + m;
      float val = acc[s][r] + bqkv[ng];
      unsigned short bv = f2bf(val);
      int b = mg / SEQ_T, t = mg % SEQ_T;
      int seg = ng / D_MODEL, rc = ng % D_MODEL;
      int h = rc / DKH, d = rc % DKH;
      int bh = b * NHEAD + h;
      if (seg == 0)      Qb[(bh * SEQ_T + t) * DKH + d] = bv;
      else if (seg == 1) Kb[(bh * SEQ_T + t) * DKH + d] = bv;
      else               Vt[(bh * DKH + d) * SEQ_T + t] = bv;
    }
  }
}

// ---------------- fused attention: scores -> softmax -> descending sort -> P*V ----------------
// block = 256 threads (8 waves); one block per (b,h, 16 even-query tile); grid = 128*24 = 3072
__global__ void k_attn(const unsigned short* __restrict__ Qb,
                       const unsigned short* __restrict__ Kb,
                       const unsigned short* __restrict__ Vt,
                       float* __restrict__ attn_out) {              // [b*384][512] f32
  extern __shared__ float probs[];                                  // 16 rows x 1024 (cols 768.. = pad)
  __shared__ float red[16][16];
  __shared__ float rowstat[16];

  const int tid  = threadIdx.x;
  const int lane = tid & 31;
  const int wv   = tid >> 5;
  const int bh   = blockIdx.x / 24;
  const int qt   = blockIdx.x % 24;
  const int m    = lane & 15, half = lane >> 4;

  const unsigned short* Qh = Qb + bh * SEQ_T * DKH;
  const unsigned short* Kh = Kb + bh * SEQ_T * DKH;
  const unsigned short* Vh = Vt + bh * DKH * SEQ_T;

  // ---- phase 1: S = Q(even) K^T / sqrt(dk), 16 x 768 into LDS ----
  {
    const unsigned short* qrow = Qh + (2 * (qt * 16 + m)) * DKH;    // even queries only
    AF A0, A1;
#pragma unroll
    for (int j = 0; j < 8; ++j) {
      int k = ((j >> 2) << 4) + half * 8 + ((j & 3) << 1);
      A0.u[j] = *(const unsigned int*)(qrow + k);
      A1.u[j] = *(const unsigned int*)(qrow + 32 + k);
    }
    for (int it = 0; it < 6; ++it) {                                // 48 n-tiles / 8 waves
      int nt = wv * 6 + it;
      const unsigned short* kcol = Kh + (nt * 16 + m) * DKH;        // B = K^T via transposed load
      AF B0, B1;
#pragma unroll
      for (int j = 0; j < 8; ++j) {
        int k = ((j >> 2) << 4) + half * 8 + ((j & 3) << 1);
        B0.u[j] = *(const unsigned int*)(kcol + k);
        B1.u[j] = *(const unsigned int*)(kcol + 32 + k);
      }
      v8f acc = {};
      acc = wmma_bf16(A0.v, B0.v, acc);
      acc = wmma_bf16(A1.v, B1.v, acc);
#pragma unroll
      for (int r = 0; r < 8; ++r)
        probs[(half * 8 + r) * 1024 + nt * 16 + m] = acc[r] * 0.125f;  // 1/sqrt(64)
    }
  }
  __syncthreads();

  // ---- phase 2: row softmax (16 threads per row) ----
  {
    const int row = tid >> 4;
    const int jj  = tid & 15;
    float* prow = probs + row * 1024;
    float lm = -1e30f;
    for (int c = jj; c < SEQ_T; c += 16) lm = fmaxf(lm, prow[c]);
    red[row][jj] = lm;
    __syncthreads();
    if (jj == 0) {
      float mx = red[row][0];
#pragma unroll
      for (int t = 1; t < 16; ++t) mx = fmaxf(mx, red[row][t]);
      rowstat[row] = mx;
    }
    __syncthreads();
    const float mx = rowstat[row];
    float ls = 0.f;
    for (int c = jj; c < SEQ_T; c += 16) {
      float e = __expf(prow[c] - mx);
      prow[c] = e;
      ls += e;
    }
    red[row][jj] = ls;
    __syncthreads();
    if (jj == 0) {
      float sm = 0.f;
#pragma unroll
      for (int t = 0; t < 16; ++t) sm += red[row][t];
      rowstat[row] = sm;
    }
    __syncthreads();
    const float inv = 1.0f / rowstat[row];
    for (int c = jj; c < SEQ_T; c += 16) prow[c] *= inv;
    for (int c = SEQ_T + jj; c < 1024; c += 16) prow[c] = -1.0f;    // pads sink below any prob
  }
  __syncthreads();

  // ---- phase 3: bitonic sort each row descending (1024 elems, 16 rows in parallel) ----
  for (int k = 2; k <= 1024; k <<= 1) {
    for (int j = k >> 1; j > 0; j >>= 1) {
      for (int p = tid; p < 16 * 512; p += 256) {
        int row = p >> 9;
        int q   = p & 511;
        int i   = ((q & ~(j - 1)) << 1) | (q & (j - 1));
        int l   = i | j;
        float* prow = probs + row * 1024;
        float a = prow[i], bq = prow[l];
        bool up = (i & k) == 0;                                     // descending overall
        if (up ? (a < bq) : (a > bq)) { prow[i] = bq; prow[l] = a; }
      }
      __syncthreads();
    }
  }

  // ---- phase 4: O = P_sorted(16x768) x V(768x64); waves 0..3 take one 16-col tile each ----
  if (wv < 4) {
    v8f acc = {};
    for (int kt = 0; kt < SEQ_T / 32; ++kt) {
      AF A;
#pragma unroll
      for (int i = 0; i < 16; ++i) {
        int j = i >> 1, pbit = i & 1;
        int k = ((j >> 2) << 4) + half * 8 + ((j & 3) << 1) + pbit;
        A.s[i] = f2bf(probs[m * 1024 + kt * 32 + k]);
      }
      const unsigned short* vcol = Vh + (wv * 16 + m) * SEQ_T + kt * 32;  // B = V via V^T layout
      AF Bf;
#pragma unroll
      for (int j = 0; j < 8; ++j)
        Bf.u[j] = *(const unsigned int*)(vcol + ((j >> 2) << 4) + half * 8 + ((j & 3) << 1));
      acc = wmma_bf16(A.v, Bf.v, acc);
    }
    int b = bh / NHEAD, h = bh % NHEAD;
#pragma unroll
    for (int r = 0; r < 8; ++r)
      attn_out[(b * T_OUT + qt * 16 + half * 8 + r) * D_MODEL + h * DKH + wv * 16 + m] = acc[r];
  }
}

// ---------------- conv branch as im2col WMMA GEMM: [6144,1536] x [1536,512] ----------------
__global__ void k_conv(const unsigned short* __restrict__ xb,
                       const unsigned short* __restrict__ wc,       // [dout][kc] bf16
                       const float* __restrict__ bconv,
                       float* __restrict__ conv_out) {              // [6144][512] f32
  const int lane = threadIdx.x & 31;
  const int wg   = blockIdx.x * 8 + (threadIdx.x >> 5);
  const int mt   = wg >> 3;          // 384 M-tiles
  const int ns   = wg & 7;           // 8 strips of 64
  const int m    = lane & 15, half = lane >> 4;
  const int mrow = mt * 16 + m;
  const int bb   = mrow / T_OUT, to = mrow % T_OUT;

  v8f acc[4] = {};
  for (int ks = 0; ks < CONV_K / 32; ++ks) {
    AF A;
#pragma unroll
    for (int j = 0; j < 8; ++j) {
      int k   = ks * 32 + ((j >> 2) << 4) + half * 8 + ((j & 3) << 1);
      int kk  = k >> 9;
      int din = k & 511;
      int t   = 2 * to - 1 + kk;                                    // stride 2, pad 1
      unsigned int val = 0u;
      if (t >= 0 && t < SEQ_T)
        val = *(const unsigned int*)(xb + (bb * SEQ_T + t) * D_MODEL + din);
      A.u[j] = val;
    }
#pragma unroll
    for (int s = 0; s < 4; ++s) {
      const unsigned short* bcol = wc + (ns * 64 + s * 16 + m) * CONV_K + ks * 32;
      AF Bf;
#pragma unroll
      for (int j = 0; j < 8; ++j)
        Bf.u[j] = *(const unsigned int*)(bcol + ((j >> 2) << 4) + half * 8 + ((j & 3) << 1));
      acc[s] = wmma_bf16(A.v, Bf.v, acc[s]);
    }
  }
#pragma unroll
  for (int s = 0; s < 4; ++s) {
#pragma unroll
    for (int r = 0; r < 8; ++r) {
      int mg = mt * 16 + half * 8 + r;
      int ng = ns * 64 + s * 16 + m;
      conv_out[mg * D_MODEL + ng] = acc[s][r] + bconv[ng];
    }
  }
}

// ---------------- residual + LayerNorm ----------------
__global__ void k_ln(const float* __restrict__ attn, const float* __restrict__ convo,
                     const float* __restrict__ gamma, const float* __restrict__ beta,
                     float* __restrict__ out) {
  __shared__ float sred[256];
  const int row = blockIdx.x;
  const int tid = threadIdx.x;
  const int c0 = tid, c1 = tid + 256;
  float a0 = attn[row * D_MODEL + c0] + convo[row * D_MODEL + c0];
  float a1 = attn[row * D_MODEL + c1] + convo[row * D_MODEL + c1];
  sred[tid] = a0 + a1;
  __syncthreads();
  for (int s = 128; s > 0; s >>= 1) { if (tid < s) sred[tid] += sred[tid + s]; __syncthreads(); }
  const float mu = sred[0] * (1.0f / 512.0f);
  __syncthreads();
  float d0 = a0 - mu, d1 = a1 - mu;
  sred[tid] = d0 * d0 + d1 * d1;
  __syncthreads();
  for (int s = 128; s > 0; s >>= 1) { if (tid < s) sred[tid] += sred[tid + s]; __syncthreads(); }
  const float rs = rsqrtf(sred[0] * (1.0f / 512.0f) + 1e-5f);
  out[row * D_MODEL + c0] = d0 * rs * gamma[c0] + beta[c0];
  out[row * D_MODEL + c1] = d1 * rs * gamma[c1] + beta[c1];
}

// ---------------- launcher ----------------
extern "C" void kernel_launch(void* const* d_in, const int* in_sizes, int n_in,
                              void* d_out, int out_size, void* d_ws, size_t ws_size,
                              hipStream_t stream) {
  const float* x     = (const float*)d_in[0];
  const float* Wqkv  = (const float*)d_in[1];
  const float* bqkv  = (const float*)d_in[2];
  const float* Wconv = (const float*)d_in[3];
  const float* bconv = (const float*)d_in[4];
  const float* gamma = (const float*)d_in[5];
  const float* beta  = (const float*)d_in[6];
  float* out = (float*)d_out;

  char* ws = (char*)d_ws;
  size_t off = 0;
  auto alloc = [&](size_t bytes) -> void* {
    void* p = ws + off;
    off = (off + bytes + 255) & ~(size_t)255;
    return p;
  };
  const size_t nX  = (size_t)BATCH * SEQ_T * D_MODEL;         // 6.29M
  const size_t nWq = (size_t)QKV_N * D_MODEL;                 // 786K
  const size_t nWc = (size_t)D_MODEL * CONV_K;                // 786K
  const size_t nQ  = (size_t)BATCH * NHEAD * SEQ_T * DKH;     // 6.29M

  unsigned short* xb  = (unsigned short*)alloc(nX * 2);
  unsigned short* wqb = (unsigned short*)alloc(nWq * 2);
  unsigned short* wcb = (unsigned short*)alloc(nWc * 2);
  unsigned short* Qb  = (unsigned short*)alloc(nQ * 2);
  unsigned short* Kb  = (unsigned short*)alloc(nQ * 2);
  unsigned short* Vt  = (unsigned short*)alloc(nQ * 2);
  float* attn  = (float*)alloc((size_t)BATCH * T_OUT * D_MODEL * 4);
  float* convo = (float*)alloc((size_t)BATCH * T_OUT * D_MODEL * 4);

  k_cvt<<<(int)((nX  + 255) / 256), 256, 0, stream>>>(x,     xb,  (int)nX);
  k_cvt<<<(int)((nWq + 255) / 256), 256, 0, stream>>>(Wqkv,  wqb, (int)nWq);
  k_cvt_wconv<<<(int)((nWc + 255) / 256), 256, 0, stream>>>(Wconv, wcb, (int)nWc);

  k_gemm_qkv<<<2304, 256, 0, stream>>>(xb, wqb, bqkv, Qb, Kb, Vt);           // 18432 waves
  k_attn<<<3072, 256, 16 * 1024 * sizeof(float), stream>>>(Qb, Kb, Vt, attn); // 64KB LDS/block
  k_conv<<<384, 256, 0, stream>>>(xb, wcb, bconv, convo);                     // 3072 waves
  k_ln<<<6144, 256, 0, stream>>>(attn, convo, gamma, beta, out);
}
